// SimpleRNN_21775484190954
// MI455X (gfx1250) — matrix operand order repaired and checked by reading, not compiled, not run on hardware
//
#include <hip/hip_runtime.h>

// ---------------------------------------------------------------------------
// tanh-RNN (B=16384, T=2048, I=1, H=20, O=1) for gfx1250 (MI455X).
//
// Transposed recurrence:  H_new^T = tanh( W_hh * H^T + C )
//   A = W_hh tiles (constant f16, A-layout), B = H^T (batch in lanes),
//   C = x[b,t]*W_ih + (b_ih+b_hh) built per-lane in C/D layout.
// D -> next-step B needs only per-lane f32->f16 packing plus a half-wave
// swap (v_permlanex16, fallback ds_swizzle SWAPX16) - no 16x16 transpose.
// ---------------------------------------------------------------------------

#define HIDDEN 20
#define TILE_B 16
#define WAVES_PER_BLOCK 8

typedef __attribute__((ext_vector_type(16))) _Float16 v16h;
typedef __attribute__((ext_vector_type(8)))  float    v8f;

static __device__ __forceinline__ float dev_tanhf(float x) {
#if defined(__AMDGCN__) && __has_builtin(__builtin_amdgcn_tanhf)
  return __builtin_amdgcn_tanhf(x);               // v_tanh_f32 (gfx1250 trans op)
#elif defined(__AMDGCN__)
  float r;
  asm("v_tanh_f32 %0, %1" : "=v"(r) : "v"(x));
  return r;
#else
  return tanhf(x);
#endif
}

static __device__ __forceinline__ unsigned pk2(float a, float b) {
  // v_cvt_pk_rtz_f16_f32
  return __builtin_bit_cast(unsigned, __builtin_amdgcn_cvt_pkrtz(a, b));
}

// Swap lane halves 0..15 <-> 16..31.
static __device__ __forceinline__ unsigned swaphalf_u(unsigned v) {
#if defined(__AMDGCN__) && __has_builtin(__builtin_amdgcn_permlanex16)
  // v_permlanex16_b32 with identity nibble selects: lane i <-> lane i+16.
  // Pure VALU: no DS counter / s_wait_dscnt on the critical path.
  return (unsigned)__builtin_amdgcn_permlanex16(
      (int)v, (int)v, (int)0x76543210, (int)0xFEDCBA98, true, false);
#else
  return (unsigned)__builtin_amdgcn_ds_swizzle((int)v, 0x401F); // SWAPX16
#endif
}
static __device__ __forceinline__ float swaphalf_f(float v) {
  return __int_as_float((int)swaphalf_u((unsigned)__float_as_int(v)));
}

__global__ __launch_bounds__(WAVES_PER_BLOCK * 32) void rnn_wmma_kernel(
    const float* __restrict__ x,     // [B, T, 1]
    const float* __restrict__ W_ih,  // [H, 1]
    const float* __restrict__ W_hh,  // [H, H]
    const float* __restrict__ b_ih,  // [H]
    const float* __restrict__ b_hh,  // [H]
    const float* __restrict__ W_out, // [1, H]
    const float* __restrict__ b_out, // [1]
    float* __restrict__ out,         // [B, 1]
    int B, int T)
{
  const int  lane = threadIdx.x & 31;
  const int  wave = threadIdx.x >> 5;
  const int  r    = lane & 15;           // batch-in-tile / row-in-tile
  const bool hi   = (lane & 16) != 0;
  const int  base = (blockIdx.x * WAVES_PER_BLOCK + wave) * TILE_B;
  const int  bidx = base + r;            // batch element this lane owns
  const int  bclr = bidx < B ? bidx : 0; // clamped for addressing

  // ---- A operands: W_hh rows j (tile1: j=0..15, tile2: j=16..19), f16
  // A k order used here (contiguous, same as B): lanes 0-15: V0..7 -> K=0..15
  //                            lanes 16-31: V0..7 -> K=16..31
  v16h A1, A2;
#pragma unroll
  for (int i = 0; i < 16; ++i) {
    const int v = i >> 1, p = i & 1;
    const int kb = (hi ? 16 : 0) + 2 * v;
    const int k  = kb + p;
    const int j2 = 16 + r;
    const float w1 = (k < HIDDEN) ? W_hh[r * HIDDEN + k] : 0.0f;
    const float w2 = (j2 < HIDDEN && k < HIDDEN) ? W_hh[j2 * HIDDEN + k] : 0.0f;
    A1[i] = (_Float16)w1;
    A2[i] = (_Float16)w2;
  }

  // ---- per-lane constants in C/D layout (VGPR m: rows j=m / j=m+8 per half)
  float wih1[8], bia1[8], wih2[4], bia2[4], wo1[8], wo2[4];
#pragma unroll
  for (int m = 0; m < 8; ++m) {
    const int j = hi ? 8 + m : m;        // tile1 rows, always < 20
    wih1[m] = W_ih[j];
    bia1[m] = b_ih[j] + b_hh[j];
    wo1[m]  = W_out[j];
  }
#pragma unroll
  for (int m = 0; m < 4; ++m) {
    const int j = 16 + m;                // tile2: only lo-half rows are real
    wih2[m] = W_ih[j];
    bia2[m] = b_ih[j] + b_hh[j];
    wo2[m]  = hi ? 0.0f : W_out[j];
  }

  // ---- state ----
  union BU { v16h v; unsigned u[8]; } Bm;   // B operand = H^T in f16 (h0 = 0)
#pragma unroll
  for (int i = 0; i < 8; ++i) Bm.u[i] = 0u;

  v8f acc1, acc2;                           // persistent WMMA accumulators
#pragma unroll
  for (int m = 0; m < 8; ++m) { acc1[m] = 0.0f; acc2[m] = 0.0f; }

  float t1[8], t2[4];                       // last step's tanh(h) for the head
#pragma unroll
  for (int m = 0; m < 8; ++m) t1[m] = 0.0f;
#pragma unroll
  for (int m = 0; m < 4; ++m) t2[m] = 0.0f;

  auto step = [&](float xv) {
    // Refresh C in-place inside the accumulators. acc2[4..7] corresponds to
    // A2's all-zero rows (j=20..23): D2[4..7]=C2[4..7], never consumed, so it
    // is left untouched (no zero-mov traffic).
#pragma unroll
    for (int m = 0; m < 8; ++m) acc1[m] = fmaf(xv, wih1[m], bia1[m]);
#pragma unroll
    for (int m = 0; m < 4; ++m) acc2[m] = fmaf(xv, wih2[m], bia2[m]);

    acc1 = __builtin_amdgcn_wmma_f32_16x16x32_f16(false, A1, false, Bm.v,
                                                  (short)0, acc1, false, false);
    acc2 = __builtin_amdgcn_wmma_f32_16x16x32_f16(false, A2, false, Bm.v,
                                                  (short)0, acc2, false, false);
#pragma unroll
    for (int m = 0; m < 8; ++m) t1[m] = dev_tanhf(acc1[m]);
#pragma unroll
    for (int m = 0; m < 4; ++m) t2[m] = dev_tanhf(acc2[m]);

    // Pack to f16 pairs, then half-swap the packed dwords that cross halves.
    const unsigned p0 = pk2(t1[0], t1[1]);  // lo: (h0,h1)   hi: (h8,h9)
    const unsigned p1 = pk2(t1[2], t1[3]);
    const unsigned p2 = pk2(t1[4], t1[5]);
    const unsigned p3 = pk2(t1[6], t1[7]);
    const unsigned q0 = pk2(t2[0], t2[1]);  // lo: (h16,h17)
    const unsigned q1 = pk2(t2[2], t2[3]);  // lo: (h18,h19)
    const unsigned sp0 = swaphalf_u(p0), sp1 = swaphalf_u(p1);
    const unsigned sp2 = swaphalf_u(p2), sp3 = swaphalf_u(p3);
    const unsigned sq0 = swaphalf_u(q0), sq1 = swaphalf_u(q1);

    // B k order used here (contiguous, same as A): lanes 0-15: Vv -> K=2v,2v+1 ; lanes 16-31: +16
    Bm.u[0] = hi ? sq0 : p0;   // hi: K=16,17  lo: K=0,1
    Bm.u[1] = hi ? sq1 : p1;   // hi: K=18,19  lo: K=2,3
    Bm.u[2] = hi ? 0u : p2;    // K>=20 zero-padded on hi half
    Bm.u[3] = hi ? 0u : p3;
    Bm.u[4] = hi ? 0u : sp0;   // lo: K=8,9 (from other half)
    Bm.u[5] = hi ? 0u : sp1;
    Bm.u[6] = hi ? 0u : sp2;
    Bm.u[7] = hi ? 0u : sp3;
  };

  // ---- time loop: 4 timesteps per float4, one-iteration double buffer ----
  const float* xrow = x + (size_t)bclr * (size_t)T;
  float4 xq = *(const float4*)xrow;
  for (int t0 = 0; t0 < T; t0 += 4) {
    const int tn = (t0 + 4 < T) ? t0 + 4 : 0;
    const float4 xn = *(const float4*)(xrow + tn);
    __builtin_prefetch(xrow + ((t0 + 256 < T) ? t0 + 256 : 0), 0, 0);
    step(xq.x);
    step(xq.y);
    step(xq.z);
    step(xq.w);
    xq = xn;
  }

  // ---- head: out[b] = sum_j h_last[b,j] * W_out[j] + b_out ----
  float partial = 0.0f;
#pragma unroll
  for (int m = 0; m < 8; ++m) partial = fmaf(t1[m], wo1[m], partial);
#pragma unroll
  for (int m = 0; m < 4; ++m) partial = fmaf(t2[m], wo2[m], partial);
  const float total = partial + swaphalf_f(partial);
  if (!hi && bidx < B) out[bidx] = total + b_out[0];
}

extern "C" void kernel_launch(void* const* d_in, const int* in_sizes, int n_in,
                              void* d_out, int out_size, void* d_ws, size_t ws_size,
                              hipStream_t stream) {
  const float* x     = (const float*)d_in[0];
  const float* W_ih  = (const float*)d_in[1];
  const float* W_hh  = (const float*)d_in[2];
  const float* b_ih  = (const float*)d_in[3];
  const float* b_hh  = (const float*)d_in[4];
  const float* W_out = (const float*)d_in[5];
  const float* b_out = (const float*)d_in[6];
  float* out = (float*)d_out;

  const int B = out_size;                  // O == 1 -> one output per batch row
  const int T = in_sizes[0] / (B > 0 ? B : 1);

  const int batchPerBlock = WAVES_PER_BLOCK * TILE_B;  // 128
  const int grid = (B + batchPerBlock - 1) / batchPerBlock;
  rnn_wmma_kernel<<<grid, WAVES_PER_BLOCK * 32, 0, stream>>>(
      x, W_ih, W_hh, b_ih, b_hh, W_out, b_out, out, B, T);
}
